// MultiHeadAttention_39943195853327
// MI455X (gfx1250) — compile-verified, matrix operand order checked by reading, not run on hardware
//
#include <hip/hip_runtime.h>
#include <hip/hip_bf16.h>
#include <math.h>

typedef __bf16 bf16_t;
typedef __attribute__((ext_vector_type(16))) __bf16 v16bf;
typedef __attribute__((ext_vector_type(8)))  __bf16 v8bf;
typedef __attribute__((ext_vector_type(4)))  __bf16 v4bf;
typedef __attribute__((ext_vector_type(8)))  float  v8f;
typedef __attribute__((ext_vector_type(4)))  int    v4i_t;

#define BATCH 4
#define NH    16
#define SEQ   1024
#define CDIM  1024
#define HD    64
#define BT    (BATCH*SEQ)   /* 4096 rows */

// Async global->LDS (CDNA5 GLOBAL_LOAD_ASYNC_TO_LDS_B128, ASYNCcnt-tracked).
#if defined(__has_builtin)
#if __has_builtin(__builtin_amdgcn_global_load_async_to_lds_b128) && \
    __has_builtin(__builtin_amdgcn_s_wait_asynccnt)
#define USE_ASYNC_LDS 1
#endif
#endif

static __device__ inline v8f vzero8f() {
  v8f z = {0.f,0.f,0.f,0.f,0.f,0.f,0.f,0.f};
  return z;
}

static __device__ inline v8f wmma_bf16(v16bf a, v16bf b, v8f c) {
  // 8-arg form: (neg_a, A, neg_b, B, c_mod, C, reuse_a, reuse_b)
  return __builtin_amdgcn_wmma_f32_16x16x32_bf16(false, a, false, b, (short)0, c, false, false);
}

// Per-lane 16-element bf16 fragment following the documented 16-bit A/B VGPR
// layout: lanes 0-15 hold K {0..7, 16..23}, lanes 16-31 hold K {8..15, 24..31}.
static __device__ inline v16bf load_frag_row(const bf16_t* __restrict__ row, int kk, int lane) {
  int h8 = (lane >> 4) * 8;
  v8bf lo = *(const v8bf*)(row + kk + h8);
  v8bf hi = *(const v8bf*)(row + kk + 16 + h8);
  v16bf r;
#pragma unroll
  for (int i = 0; i < 8; ++i) { r[i] = lo[i]; r[8 + i] = hi[i]; }
  return r;
}

static __device__ inline v16bf frag_from_chunks(v8bf lo, v8bf hi) {
  v16bf r;
#pragma unroll
  for (int i = 0; i < 8; ++i) { r[i] = lo[i]; r[8 + i] = hi[i]; }
  return r;
}

// Stage one 16B chunk from global into LDS. Async path goes straight to LDS
// without touching VGPRs; fallback relays through registers.
static __device__ inline void stage_chunk(const bf16_t* g, bf16_t* l) {
#ifdef USE_ASYNC_LDS
  __builtin_amdgcn_global_load_async_to_lds_b128(
      (__attribute__((address_space(1))) v4i_t*)g,
      (__attribute__((address_space(3))) v4i_t*)l, 0, 0);
#else
  *(v8bf*)l = *(const v8bf*)g;
#endif
}

static __device__ inline void stage_row32(const bf16_t* g, bf16_t* l) {
  stage_chunk(g,      l);
  stage_chunk(g + 8,  l + 8);
  stage_chunk(g + 16, l + 16);
  stage_chunk(g + 24, l + 24);
}

static __device__ inline void async_wait_all() {
#ifdef USE_ASYNC_LDS
  __builtin_amdgcn_s_wait_asynccnt(0);
#endif
}

// ---------------------------------------------------------------------------
// fp32 -> bf16 conversion (vectorized 4 elems/thread)
// ---------------------------------------------------------------------------
__global__ __launch_bounds__(256) void cvt_f32_bf16_x4(
    const float4* __restrict__ src, v4bf* __restrict__ dst, int n4) {
  int i = blockIdx.x * blockDim.x + threadIdx.x;
  if (i < n4) {
    float4 f = src[i];
    v4bf o;
    o[0] = (bf16_t)f.x; o[1] = (bf16_t)f.y; o[2] = (bf16_t)f.z; o[3] = (bf16_t)f.w;
    dst[i] = o;
  }
}

// ---------------------------------------------------------------------------
// GEMM: out[m,n] = sum_k A[m,k] * W[n,k]   (nn.Linear: y = x @ W^T)
// Double-buffered async-LDS staging: per 32-wide k-step the block stages
// A(32x32) + W(256x32) into LDS (padded rows), 8 waves compute 16x64 tiles
// from LDS fragments. Async copies overlap the WMMA stream; barrier sits
// after compute so the buffer being read is never overwritten.
// mode 0: bf16 [B,NH,SEQ,HD] (head-split Q/K)
// mode 1: fp32 row-major [M,N] (final projection)
// mode 2: bf16 [B,NH,HD,SEQ] (transposed V)
// ---------------------------------------------------------------------------
#define BM 32
#define BN 256
#define BK 32
#define WPAD 40   /* 80B rows: 16-lane b128 frag reads hit disjoint bank blocks */

__global__ __launch_bounds__(256) void gemm_bf16_wmma(
    const bf16_t* __restrict__ A, const bf16_t* __restrict__ W,
    void* __restrict__ out, int M, int N, int K, int mode) {
  __shared__ bf16_t Asm_[2][BM][WPAD];
  __shared__ bf16_t Wsm_[2][BN][WPAD];

  int tid = threadIdx.x;
  int lane = tid & 31;
  int wave = tid >> 5;
  int nBlocks = N / BN;
  int bm = blockIdx.x / nBlocks;
  int bn = blockIdx.x % nBlocks;
  int m0 = bm * BM, n0 = bn * BN;

  const bf16_t* wsrc = W + (size_t)(n0 + tid) * K;                 // one W row/thread
  const bf16_t* asrc = A + (size_t)(m0 + (tid & (BM - 1))) * K;    // threads 0..31

  v8f acc[4];
#pragma unroll
  for (int t = 0; t < 4; ++t) acc[t] = vzero8f();

  // prologue: stage k-slice 0 into buffer 0
  stage_row32(wsrc, &Wsm_[0][tid][0]);
  if (tid < BM) stage_row32(asrc, &Asm_[0][tid][0]);
  async_wait_all();
  __syncthreads();

  int wm = (wave >> 2) * 16;
  int wn = (wave & 3) * 64;
  int r = lane & 15;
  int h8 = (lane >> 4) * 8;

  int nsteps = K / BK;
  int buf = 0;
  for (int s = 0; s < nsteps; ++s) {
    if (s + 1 < nsteps) {                    // stage next slice into other buf
      int k0 = (s + 1) * BK;
      stage_row32(wsrc + k0, &Wsm_[buf ^ 1][tid][0]);
      if (tid < BM) stage_row32(asrc + k0, &Asm_[buf ^ 1][tid][0]);
    }
    // compute from current buf (ds_load_b128 fragments)
    const bf16_t* ar = &Asm_[buf][wm + r][0];
    v16bf af = frag_from_chunks(*(const v8bf*)(ar + h8),
                                *(const v8bf*)(ar + 16 + h8));
#pragma unroll
    for (int t = 0; t < 4; ++t) {
      const bf16_t* wr = &Wsm_[buf][wn + t * 16 + r][0];
      v16bf bfv = frag_from_chunks(*(const v8bf*)(wr + h8),
                                   *(const v8bf*)(wr + 16 + h8));
      acc[t] = wmma_bf16(af, bfv, acc[t]);
    }
    async_wait_all();      // next-stage loads landed during the WMMAs
    __syncthreads();       // all waves done reading buf -> safe to overwrite
    buf ^= 1;
  }

  int hh = lane >> 4, nn = lane & 15;
  if (mode == 0) {
    bf16_t* q = (bf16_t*)out;
#pragma unroll
    for (int t = 0; t < 4; ++t) {
      int n = n0 + wn + t * 16 + nn;
      int head = n >> 6, d = n & (HD - 1);
#pragma unroll
      for (int rr = 0; rr < 8; ++rr) {
        int m = m0 + wm + rr + 8 * hh;       // m = b*SEQ + t
        int b = m >> 10, tt = m & (SEQ - 1);
        q[(((size_t)(b * NH + head) * SEQ) + tt) * HD + d] = (bf16_t)acc[t][rr];
      }
    }
  } else if (mode == 2) {
    bf16_t* vt = (bf16_t*)out;
#pragma unroll
    for (int t = 0; t < 4; ++t) {
      int n = n0 + wn + t * 16 + nn;
      int head = n >> 6, d = n & (HD - 1);
#pragma unroll
      for (int rr = 0; rr < 8; ++rr) {
        int m = m0 + wm + rr + 8 * hh;
        int b = m >> 10, tt = m & (SEQ - 1);
        vt[(((size_t)(b * NH + head) * HD) + d) * SEQ + tt] = (bf16_t)acc[t][rr];
      }
    }
  } else {
    float* o = (float*)out;
#pragma unroll
    for (int t = 0; t < 4; ++t) {
      int n = n0 + wn + t * 16 + nn;
#pragma unroll
      for (int rr = 0; rr < 8; ++rr) {
        int m = m0 + wm + rr + 8 * hh;
        o[(size_t)m * N + n] = acc[t][rr];
      }
    }
  }
}

// ---------------------------------------------------------------------------
// Attention: per block = (b, h, 16-query tile), 8 wave32s.
// Phase 1: S = Q K^T * scale + alibi; full 1024-wide rows in LDS (fp32).
// Phase 2: row softmax (float4-vectorized, shfl reductions).
// Phase 3: coalesced float4 attn stores + pack bf16 P to LDS.
// Phase 4: Y = P V via WMMA; P frags = ds_load_b128 pairs, V frags =
//          contiguous b128s from transposed V. Split-K across wave pairs.
// ---------------------------------------------------------------------------
#define QT 16
#define SPAD 1028   /* fp32 row stride: 1028 % 64 == 4 -> rows 4 banks apart */
#define PPAD 1032   /* bf16 row stride: 2064B = 516 dwords -> 4 banks apart */

__global__ __launch_bounds__(256) void attn_wmma(
    const bf16_t* __restrict__ Qm, const bf16_t* __restrict__ Km,
    const bf16_t* __restrict__ Vt, float* __restrict__ attn_out,
    bf16_t* __restrict__ Yb) {
  __shared__ float  S[QT][SPAD];
  __shared__ bf16_t P[QT][PPAD];

  int lane = threadIdx.x & 31;
  int wave = threadIdx.x >> 5;
  int blk = blockIdx.x;                 // [0, B*NH*(SEQ/QT))
  int qt = blk & 63;
  int bh = blk >> 6;                    // b*NH + h
  int h = bh & (NH - 1);
  int b = bh >> 4;
  int q0 = qt * QT;
  float slope = exp2f(-0.5f * (float)(h + 1));  // 2^(-8*(h+1)/NH), NH=16

  const bf16_t* Qbase = Qm + (size_t)bh * SEQ * HD;
  const bf16_t* Kbase = Km + (size_t)bh * SEQ * HD;

  // ---- Phase 1: scores into LDS --------------------------------------
  int mrow = lane & 15;
  int hh = lane >> 4;
  const bf16_t* qrow = Qbase + (size_t)(q0 + mrow) * HD;
  v16bf qa0 = load_frag_row(qrow, 0, lane);
  v16bf qa1 = load_frag_row(qrow, 32, lane);

  int c0 = wave * 128;
  for (int t = 0; t < 8; ++t) {
    int n0 = c0 + t * 16;
    const bf16_t* krow = Kbase + (size_t)(n0 + mrow) * HD;  // lane's key row
    v16bf kb0 = load_frag_row(krow, 0, lane);
    v16bf kb1 = load_frag_row(krow, 32, lane);
    v8f acc = vzero8f();
    acc = wmma_bf16(qa0, kb0, acc);
    acc = wmma_bf16(qa1, kb1, acc);
    int n = n0 + (lane & 15);
#pragma unroll
    for (int r = 0; r < 8; ++r) {
      int mm = r + 8 * hh;
      float dq = (float)(q0 + mm) - (float)n;
      S[mm][n] = acc[r] * 0.125f - slope * fabsf(dq);
    }
  }
  __syncthreads();

  // ---- Phase 2/3: softmax + attn output + bf16 P ---------------------
  int row = threadIdx.x >> 4;
  int sub = threadIdx.x & 15;
  float4* Srow4 = (float4*)&S[row][0];  // row base 4112B -> 16B aligned

  float mx = -3.0e38f;
#pragma unroll
  for (int jj = 0; jj < 16; ++jj) {
    float4 v = Srow4[sub + jj * 16];
    mx = fmaxf(mx, fmaxf(fmaxf(v.x, v.y), fmaxf(v.z, v.w)));
  }
#pragma unroll
  for (int o = 8; o; o >>= 1) mx = fmaxf(mx, __shfl_xor(mx, o, 16));

  float sum = 0.f;
#pragma unroll
  for (int jj = 0; jj < 16; ++jj) {
    int idx = sub + jj * 16;
    float4 v = Srow4[idx];
    v.x = __expf(v.x - mx); v.y = __expf(v.y - mx);
    v.z = __expf(v.z - mx); v.w = __expf(v.w - mx);
    Srow4[idx] = v;
    sum += (v.x + v.y) + (v.z + v.w);
  }
#pragma unroll
  for (int o = 8; o; o >>= 1) sum += __shfl_xor(sum, o, 16);
  float inv = 1.0f / sum;

  float4* arow4 = (float4*)(attn_out + ((size_t)bh * SEQ + (q0 + row)) * SEQ);
#pragma unroll
  for (int jj = 0; jj < 16; ++jj) {
    int idx = sub + jj * 16;
    float4 v = Srow4[idx];
    float4 p;
    p.x = v.x * inv; p.y = v.y * inv; p.z = v.z * inv; p.w = v.w * inv;
    arow4[idx] = p;                     // coalesced global_store_b128
    v4bf pb;
    pb[0] = (bf16_t)p.x; pb[1] = (bf16_t)p.y;
    pb[2] = (bf16_t)p.z; pb[3] = (bf16_t)p.w;
    *(v4bf*)&P[row][idx * 4] = pb;      // 8B ds store
  }
  __syncthreads();

  // ---- Phase 4: Y = P V (split-K across wave pairs) ------------------
  int d0 = (wave & 3) * 16;
  int kb0 = (wave >> 2) * 512;
  int nlane = lane & 15;
  int hh8 = hh * 8;
  const bf16_t* vtrow = Vt + ((size_t)bh * HD + d0 + nlane) * SEQ;

  v8f acc = vzero8f();
  for (int kb = kb0; kb < kb0 + 512; kb += 32) {
    v16bf pa = frag_from_chunks(*(const v8bf*)&P[nlane][kb + hh8],
                                *(const v8bf*)&P[nlane][kb + 16 + hh8]);
    v16bf vb = load_frag_row(vtrow, kb, lane);           // contiguous b128s
    acc = wmma_bf16(pa, vb, acc);
  }

  // split-K reduction through (now dead) S region
  float* red = &S[0][0];
  if (wave >= 4) {
#pragma unroll
    for (int r = 0; r < 8; ++r)
      red[(wave - 4) * 256 + (r + 8 * hh) * 16 + (lane & 15)] = acc[r];
  }
  __syncthreads();
  if (wave < 4) {
#pragma unroll
    for (int r = 0; r < 8; ++r) {
      int mm = r + 8 * hh;
      float v = acc[r] + red[wave * 256 + mm * 16 + (lane & 15)];
      int tpos = q0 + mm;
      Yb[((size_t)(b * SEQ + tpos)) * CDIM + h * HD + d0 + (lane & 15)] = (bf16_t)v;
    }
  }
}

// ---------------------------------------------------------------------------
extern "C" void kernel_launch(void* const* d_in, const int* in_sizes, int n_in,
                              void* d_out, int out_size, void* d_ws, size_t ws_size,
                              hipStream_t stream) {
  const float* x  = (const float*)d_in[0];
  const float* Wq = (const float*)d_in[1];
  const float* Wk = (const float*)d_in[2];
  const float* Wv = (const float*)d_in[3];
  const float* Wp = (const float*)d_in[4];

  const size_t XN = (size_t)BT * CDIM;       // 4M
  const size_t WN = (size_t)CDIM * CDIM;     // 1M

  bf16_t* xb  = (bf16_t*)d_ws;
  bf16_t* Wqb = xb + XN;
  bf16_t* Wkb = Wqb + WN;
  bf16_t* Wvb = Wkb + WN;
  bf16_t* Wpb = Wvb + WN;
  bf16_t* Qb  = Wpb + WN;
  bf16_t* Kb  = Qb + XN;
  bf16_t* Vtb = Kb + XN;                     // [B,NH,HD,SEQ] transposed
  bf16_t* Yb  = Vtb + XN;                    // total 24M bf16 = 48 MB

  float* yout = (float*)d_out;
  float* attn = yout + XN;                   // (y, attn) concatenated

  // fp32 -> bf16
  cvt_f32_bf16_x4<<<(int)(XN / 4 / 256), 256, 0, stream>>>((const float4*)x,  (v4bf*)xb,  (int)(XN / 4));
  cvt_f32_bf16_x4<<<(int)(WN / 4 / 256), 256, 0, stream>>>((const float4*)Wq, (v4bf*)Wqb, (int)(WN / 4));
  cvt_f32_bf16_x4<<<(int)(WN / 4 / 256), 256, 0, stream>>>((const float4*)Wk, (v4bf*)Wkb, (int)(WN / 4));
  cvt_f32_bf16_x4<<<(int)(WN / 4 / 256), 256, 0, stream>>>((const float4*)Wv, (v4bf*)Wvb, (int)(WN / 4));
  cvt_f32_bf16_x4<<<(int)(WN / 4 / 256), 256, 0, stream>>>((const float4*)Wp, (v4bf*)Wpb, (int)(WN / 4));

  // Projections: Q/K head-split, V transposed for the P*V B-fragment path
  int gemmGrid = (BT / 32) * (CDIM / 256);   // 512 blocks
  gemm_bf16_wmma<<<gemmGrid, 256, 0, stream>>>(xb, Wqb, Qb,  BT, CDIM, CDIM, 0);
  gemm_bf16_wmma<<<gemmGrid, 256, 0, stream>>>(xb, Wkb, Kb,  BT, CDIM, CDIM, 0);
  gemm_bf16_wmma<<<gemmGrid, 256, 0, stream>>>(xb, Wvb, Vtb, BT, CDIM, CDIM, 2);

  // fused scores + softmax + attn output + P*V
  attn_wmma<<<BATCH * NH * (SEQ / QT), 256, 0, stream>>>(Qb, Kb, Vtb, attn, Yb);

  // output projection -> fp32 y
  gemm_bf16_wmma<<<gemmGrid, 256, 0, stream>>>(Yb, Wpb, yout, BT, CDIM, CDIM, 1);
}
